// RWKVv7Separator_2594160247317
// MI455X (gfx1250) — matrix operand order repaired and checked by reading, not compiled, not run on hardware
//
#include <hip/hip_runtime.h>

// ---------------------------------------------------------------------------
// RWKV-v7 bidirectional separator forward for MI455X (gfx1250, wave32, WMMA)
// All GEMMs run on v_wmma_f32_16x16x32_bf16 (bf16 in, f32 accum).
// ---------------------------------------------------------------------------

typedef __bf16 bf16_t;
typedef __attribute__((ext_vector_type(16))) __bf16 v16bf;
typedef __attribute__((ext_vector_type(8)))  __bf16 v8bf;
typedef __attribute__((ext_vector_type(8)))  float  v8f;

// Fixed problem dims (from the reference setup_inputs)
constexpr int Bc = 4, Tc = 2048, Cc = 512, Hd = 256, Lc = 6, nHd = 4, HSc = 64;
constexpr int Dn = 2;                         // directions (fwd, bwd)
constexpr long MT  = (long)Bc * Tc;           // 8192 tokens per direction
constexpr long NAd = MT * Hd;                 // per-direction activation elems
constexpr float GN_EPS_C = 1e-5f * 64.0f * 64.0f;

// -------------------------------- WMMA GEMM --------------------------------

__device__ __forceinline__ v16bf ld_fragA(const bf16_t* p) {
  // 16-bit A 16x32 layout: per-lane K chunks [k0+half*8 .. +8) and [+16 .. +24)
  v8bf lo = *(const v8bf*)p;
  v8bf hi = *(const v8bf*)(p + 16);
  v16bf f;
#pragma unroll
  for (int i = 0; i < 8; ++i) { f[i] = lo[i]; f[i + 8] = hi[i]; }
  return f;
}

__device__ __forceinline__ v16bf ld_fragB(const bf16_t* p) {
  // 16-bit B 32x16 layout (from N-major weights): per-lane K chunks [k0+half*8 .. +8) and [+16 .. +24)
  v8bf lo = *(const v8bf*)p;
  v8bf hi = *(const v8bf*)(p + 16);
  v16bf f;
#pragma unroll
  for (int i = 0; i < 8; ++i) { f[i] = lo[i]; f[i + 8] = hi[i]; }
  return f;
}

// EPI: 0=none 1=sigmoid 2=tanh 3=relu^2 4=snake ; bias (optional) added first.
template <int EPI, bool ACC, typename OutT>
__global__ __launch_bounds__(128) void gemm_bf16_kernel(
    const bf16_t* __restrict__ A,   // (batch, M, K) row major, bf16
    const bf16_t* __restrict__ Bt,  // (batch, N, K) row major, bf16 (B transposed)
    const float*  __restrict__ bias,
    OutT* __restrict__ Out,         // (batch, M, N)
    int M, int N, int K, int WN,
    long aStride, long bStride, long biasStride, long oStride,
    const float* __restrict__ alphaPtr)
{
  const int bz = blockIdx.z;
  A  += (size_t)bz * aStride;
  Bt += (size_t)bz * bStride;
  Out += (size_t)bz * oStride;
  const float* bp = bias ? bias + (size_t)bz * biasStride : nullptr;

  const int tid  = threadIdx.x;
  const int wave = tid >> 5, lane = tid & 31;
  const int wm = wave / WN, wn = wave % WN;
  const int BMw = (4 / WN) * 32, BNw = WN * 32;
  const int mTile = blockIdx.y * BMw + wm * 32;
  const int nTile = blockIdx.x * BNw + wn * 32;
  const int lrow = lane & 15, half = lane >> 4;

  v8f acc[2][2] = {};

  const bf16_t* a0p = A  + (size_t)(mTile + lrow) * K;
  const bf16_t* a1p = A  + (size_t)(mTile + 16 + lrow) * K;
  const bf16_t* b0p = Bt + (size_t)(nTile + lrow) * K;
  const bf16_t* b1p = Bt + (size_t)(nTile + 16 + lrow) * K;

  for (int k0 = 0; k0 < K; k0 += 32) {
    v16bf af0 = ld_fragA(a0p + k0 + half * 8);
    v16bf af1 = ld_fragA(a1p + k0 + half * 8);
    v16bf bf0 = ld_fragB(b0p + k0 + half * 8);
    v16bf bf1 = ld_fragB(b1p + k0 + half * 8);
    acc[0][0] = __builtin_amdgcn_wmma_f32_16x16x32_bf16(false, af0, false, bf0, (short)0, acc[0][0], false, false);
    acc[0][1] = __builtin_amdgcn_wmma_f32_16x16x32_bf16(false, af0, false, bf1, (short)0, acc[0][1], false, false);
    acc[1][0] = __builtin_amdgcn_wmma_f32_16x16x32_bf16(false, af1, false, bf0, (short)0, acc[1][0], false, false);
    acc[1][1] = __builtin_amdgcn_wmma_f32_16x16x32_bf16(false, af1, false, bf1, (short)0, acc[1][1], false, false);
  }

  float alpha = 1.0f;
  if (EPI == 4) alpha = alphaPtr ? *alphaPtr : 1.0f;

#pragma unroll
  for (int i = 0; i < 2; ++i) {
#pragma unroll
    for (int j = 0; j < 2; ++j) {
      const int rowB = mTile + i * 16 + half * 8;
      const int col  = nTile + j * 16 + lrow;
      const float bv = bp ? bp[col] : 0.0f;
#pragma unroll
      for (int rr = 0; rr < 8; ++rr) {
        float v = acc[i][j][rr] + bv;
        if (EPI == 1)      v = 1.0f / (1.0f + expf(-v));
        else if (EPI == 2) v = tanhf(v);
        else if (EPI == 3) { float t = fmaxf(v, 0.0f); v = t * t; }
        else if (EPI == 4) { float s = sinf(alpha * v); v += s * s / alpha; }
        size_t oi = (size_t)(rowB + rr) * (size_t)N + col;
        if (ACC) v += ((float*)Out)[oi];
        Out[oi] = (OutT)v;
      }
    }
  }
}

// ----------------------------- conversion kernels --------------------------

__global__ __launch_bounds__(256) void conv_tr_kernel(
    const float* __restrict__ src, bf16_t* __restrict__ dst, int K, int N)
{
  long b = blockIdx.y;
  src += (size_t)b * K * N;
  dst += (size_t)b * K * N;
  long tot = (long)K * N;
  for (long idx = (long)blockIdx.x * 256 + threadIdx.x; idx < tot;
       idx += (long)gridDim.x * 256) {
    long n = idx / K, kk = idx % K;
    dst[idx] = (bf16_t)src[kk * (long)N + n];
  }
}

__global__ __launch_bounds__(256) void conv_bf_kernel(
    const float* __restrict__ s, bf16_t* __restrict__ d, long n)
{
  long i = (long)blockIdx.x * 256 + threadIdx.x;
  if (i < n) d[i] = (bf16_t)s[i];
}

// ------------------------------- LayerNorm ---------------------------------

template <typename OutT>
__global__ __launch_bounds__(256) void ln_kernel(
    const float* __restrict__ x, const float* __restrict__ w,
    const float* __restrict__ b, OutT* __restrict__ out,
    int C, int tokensPerDir, long pStride, float eps)
{
  __shared__ float red[256];
  const int tok = blockIdx.x, tid = threadIdx.x;
  const int d = tok / tokensPerDir;
  const float* wp = w + (size_t)d * pStride;
  const float* bp = b + (size_t)d * pStride;
  const float* xp = x + (size_t)tok * C;
  OutT* op = out + (size_t)tok * C;

  float s = 0.0f;
  for (int i = tid; i < C; i += 256) s += xp[i];
  red[tid] = s; __syncthreads();
#pragma unroll
  for (int off = 128; off > 0; off >>= 1) {
    if (tid < off) red[tid] += red[tid + off];
    __syncthreads();
  }
  const float mu = red[0] / (float)C;
  __syncthreads();
  s = 0.0f;
  for (int i = tid; i < C; i += 256) { float dd = xp[i] - mu; s += dd * dd; }
  red[tid] = s; __syncthreads();
#pragma unroll
  for (int off = 128; off > 0; off >>= 1) {
    if (tid < off) red[tid] += red[tid + off];
    __syncthreads();
  }
  const float rs = rsqrtf(red[0] / (float)C + eps);
  for (int i = tid; i < C; i += 256)
    op[i] = (OutT)((xp[i] - mu) * rs * wp[i] + bp[i]);
}

// --------------------------- token-shift mixers ----------------------------

__global__ __launch_bounds__(256) void mix6_kernel(
    const float* __restrict__ xn,
    const float* __restrict__ wr, const float* __restrict__ ww,
    const float* __restrict__ wk, const float* __restrict__ wv,
    const float* __restrict__ wa, const float* __restrict__ wg,
    long pStride,
    bf16_t* __restrict__ oR, bf16_t* __restrict__ oW, bf16_t* __restrict__ oK,
    bf16_t* __restrict__ oV, bf16_t* __restrict__ oA, bf16_t* __restrict__ oG)
{
  const int tok = blockIdx.x, i = threadIdx.x;
  const int d = tok / (Bc * Tc);
  const int t = tok % Tc;
  const size_t gi = (size_t)tok * Hd + i;
  const float xc = xn[gi];
  float xp = 0.0f;
  if (d == 0) { if (t > 0) xp = xn[gi - Hd]; }
  else        { if (t < Tc - 1) xp = xn[gi + Hd]; }
  const float xx = xp - xc;
  const size_t po = (size_t)d * pStride + i;
  oR[gi] = (bf16_t)(xc + xx * wr[po]);
  oW[gi] = (bf16_t)(xc + xx * ww[po]);
  oK[gi] = (bf16_t)(xc + xx * wk[po]);
  oV[gi] = (bf16_t)(xc + xx * wv[po]);
  oA[gi] = (bf16_t)(xc + xx * wa[po]);
  oG[gi] = (bf16_t)(xc + xx * wg[po]);
}

__global__ __launch_bounds__(256) void cmix_kernel(
    const float* __restrict__ xn, const float* __restrict__ cxk,
    long pStride, bf16_t* __restrict__ oK)
{
  const int tok = blockIdx.x, i = threadIdx.x;
  const int d = tok / (Bc * Tc);
  const int t = tok % Tc;
  const size_t gi = (size_t)tok * Hd + i;
  const float xc = xn[gi];
  float xp = 0.0f;
  if (d == 0) { if (t > 0) xp = xn[gi - Hd]; }
  else        { if (t < Tc - 1) xp = xn[gi + Hd]; }
  oK[gi] = (bf16_t)(xc + (xp - xc) * cxk[(size_t)d * pStride + i]);
}

// ------------------------- pre-scan vector math ----------------------------

__device__ __forceinline__ float head_sum(float* red, int i, float val) {
  __syncthreads();
  red[i] = val;
  __syncthreads();
#pragma unroll
  for (int off = 32; off > 0; off >>= 1) {
    if ((i & 63) < off) red[i] += red[i + off];
    __syncthreads();
  }
  return red[i & ~63];
}

__global__ __launch_bounds__(256) void vecops_kernel(
    float* __restrict__ k, float* __restrict__ v,
    const float* __restrict__ aB, float* __restrict__ wraw,
    const float* __restrict__ vgate, float* __restrict__ vfirst,
    float* __restrict__ a4, float* __restrict__ b4,
    const float* __restrict__ kkw, const float* __restrict__ kaw,
    long pStride, int lidx)
{
  __shared__ float red[256];
  const int tok = blockIdx.x, i = threadIdx.x;
  const int d = tok / (Bc * Tc);
  const size_t gi = (size_t)tok * Hd + i;
  const size_t po = (size_t)d * pStride + i;

  const float kvv = k[gi];
  const float kk  = kvv * kkw[po];
  float nrm = sqrtf(head_sum(red, i, kk * kk));
  nrm = fmaxf(nrm, 1e-12f);
  const float kkn = kk / nrm;
  const float av  = aB[gi];
  a4[gi] = -kkn;
  b4[gi] = kkn * av;
  k[gi]  = kvv * (1.0f + (av - 1.0f) * kaw[po]);

  const float wr = wraw[gi];
  const float wv = -log1pf(expf(-wr)) - 0.5f;   // -softplus(-w) - 0.5
  wraw[gi] = expf(-expf(wv));                   // decay in (0,1)

  const float vv = v[gi];
  if (lidx == 0) vfirst[gi] = vv;
  else           v[gi] = vv + (vfirst[gi] - vv) * vgate[gi];
}

// ------------------------------- WKV scan ----------------------------------

__global__ __launch_bounds__(64) void wkv_scan_kernel(
    const float* __restrict__ r, const float* __restrict__ wdec,
    const float* __restrict__ k, const float* __restrict__ v,
    const float* __restrict__ a4, const float* __restrict__ b4,
    float* __restrict__ o)
{
  __shared__ float rv[64], wv[64], kv[64], av[64], bv[64];
  const int blk = blockIdx.x;            // d*B*nH + b*nH + h
  const int h = blk % nHd;
  const int b = (blk / nHd) % Bc;
  const int d = blk / (nHd * Bc);
  const int i = threadIdx.x;
  const size_t base = (((size_t)d * Bc + b) * Tc) * Hd + (size_t)h * HSc;

  float S[64];
#pragma unroll
  for (int j = 0; j < 64; ++j) S[j] = 0.0f;

  for (int st = 0; st < Tc; ++st) {
    const int t = (d == 0) ? st : (Tc - 1 - st);
    const size_t off = base + (size_t)t * Hd;
    __syncthreads();
    rv[i] = r[off + i]; wv[i] = wdec[off + i]; kv[i] = k[off + i];
    av[i] = a4[off + i]; bv[i] = b4[off + i];
    const float vi = v[off + i];
    if (st + 1 < Tc) {   // prefetch next timestep (global_prefetch)
      const int tn = (d == 0) ? (st + 1) : (Tc - 2 - st);
      const size_t offn = base + (size_t)tn * Hd + i;
      __builtin_prefetch(r + offn, 0, 0);
      __builtin_prefetch(k + offn, 0, 0);
      __builtin_prefetch(wdec + offn, 0, 0);
    }
    __syncthreads();
    float sa = 0.0f;
#pragma unroll
    for (int j = 0; j < 64; ++j) sa += S[j] * av[j];
    float y = 0.0f;
#pragma unroll
    for (int j = 0; j < 64; ++j) {
      S[j] = S[j] * wv[j] + sa * bv[j] + vi * kv[j];
      y += S[j] * rv[j];
    }
    o[off + i] = y;
  }
}

// ------------------------ post-scan GroupNorm + bonus ----------------------

__global__ __launch_bounds__(256) void postgn_kernel(
    const float* __restrict__ o, const float* __restrict__ r,
    const float* __restrict__ k, const float* __restrict__ v,
    const float* __restrict__ g,
    const float* __restrict__ lnxw, const float* __restrict__ lnxb,
    const float* __restrict__ rkw, long pStride, bf16_t* __restrict__ og)
{
  __shared__ float red[256];
  const int tok = blockIdx.x, i = threadIdx.x;
  const int d = tok / (Bc * Tc);
  const size_t gi = (size_t)tok * Hd + i;
  const size_t po = (size_t)d * pStride + i;

  const float ov = o[gi];
  const float mu = head_sum(red, i, ov) * (1.0f / HSc);
  const float dd = ov - mu;
  const float var = head_sum(red, i, dd * dd) * (1.0f / HSc);
  const float gn = dd * rsqrtf(var + GN_EPS_C) * lnxw[po] + lnxb[po];
  const float dot = head_sum(red, i, r[gi] * k[gi] * rkw[po]);
  og[gi] = (bf16_t)((gn + dot * v[gi]) * g[gi]);
}

// ------------------------------ head helpers -------------------------------

__global__ __launch_bounds__(256) void hcombine_kernel(
    const float* __restrict__ x, bf16_t* __restrict__ hbf)
{
  const size_t i = (size_t)blockIdx.x * 256 + threadIdx.x;
  hbf[i] = (bf16_t)(0.5f * (x[i] + x[i + NAd]));
}

__global__ __launch_bounds__(256) void final_kernel(
    const float* __restrict__ z, const float* __restrict__ logits,
    const float* __restrict__ r1, const float* __restrict__ r2,
    float* __restrict__ out)
{
  const size_t gi = (size_t)blockIdx.x * 256 + threadIdx.x;
  const size_t tok = gi / Cc;
  const int c = (int)(gi % Cc);
  const float l0 = logits[tok * (2 * Cc) + 2 * c];
  const float l1 = logits[tok * (2 * Cc) + 2 * c + 1];
  const float mx = fmaxf(l0, l1);
  const float e0 = expf(l0 - mx), e1 = expf(l1 - mx);
  const float inv = 1.0f / (e0 + e1);
  const float m1 = e0 * inv, m2 = e1 * inv;
  const float zz = z[gi], a = r1[gi], bb = r2[gi];
  const size_t P = (size_t)MT * Cc;
  out[gi]         = m1 * zz + a;
  out[P + gi]     = m2 * zz + bb;
  out[2 * P + gi] = m1;
  out[3 * P + gi] = m2;
  out[4 * P + gi] = a;
  out[5 * P + gi] = bb;
}

// ------------------------------ host dispatch ------------------------------

static void gemm(hipStream_t st, const bf16_t* A, const bf16_t* Bt,
                 const float* bias, void* out, bool outBf, int epi, bool acc,
                 int M, int N, int K, int batch,
                 long aS, long bS, long biasS, long oS,
                 const float* alpha = nullptr)
{
  const int WN = (N >= 64) ? 2 : 1;
  dim3 grid(N / (WN * 32), M / ((4 / WN) * 32), batch), blk(128);
#define G_CALL(E, A_, T_)                                                    \
  gemm_bf16_kernel<E, A_, T_><<<grid, blk, 0, st>>>(                         \
      A, Bt, bias, (T_*)out, M, N, K, WN, aS, bS, biasS, oS, alpha)
  if (acc) { G_CALL(0, true, float); return; }
  if (outBf) {
    switch (epi) {
      case 0: G_CALL(0, false, bf16_t); break;
      case 1: G_CALL(1, false, bf16_t); break;
      case 2: G_CALL(2, false, bf16_t); break;
      case 3: G_CALL(3, false, bf16_t); break;
      default: G_CALL(4, false, bf16_t); break;
    }
  } else {
    switch (epi) {
      case 0: G_CALL(0, false, float); break;
      default: G_CALL(1, false, float); break;
    }
  }
#undef G_CALL
}

// Input order: z_mix first, then params pytree leaves in sorted-key order.
enum {
  IN_ZMIX = 0,
  C_W_k, C_W_o, C_W_r, C_W_v,
  C_a0, C_a1, C_a2,
  C_cWk, C_cWv, C_c_xk,
  C_g1, C_g2,
  C_k_a, C_k_k,
  C_ln1_b, C_ln1_w, C_ln2_b, C_ln2_w,
  C_lnx_b, C_lnx_w,
  C_r_k,
  C_v0, C_v1, C_v2,
  C_w0, C_w1, C_w2,
  C_x_a, C_x_g, C_x_k, C_x_r, C_x_v, C_x_w,
  P_down_W, P_down_b,
  P_hm_W, P_hm_b, P_hm_ln_b, P_hm_ln_w,
  H1_W1, H1_W2, H1_alpha, H1_b1, H1_b2, H1_ln_b, H1_ln_w,
  H2_W1, H2_W2, H2_alpha, H2_b1, H2_b2, H2_ln_b, H2_ln_w,
  P_up_W, P_up_b, IN_COUNT
};

extern "C" void kernel_launch(void* const* d_in, const int* in_sizes, int n_in,
                              void* d_out, int out_size, void* d_ws,
                              size_t ws_size, hipStream_t stream)
{
  (void)in_sizes; (void)n_in; (void)out_size; (void)ws_size;
#define IN_F(i) ((const float*)d_in[(i)])
  const float* zmix = IN_F(IN_ZMIX);

  // --- workspace bump allocator (256B aligned) ---
  char* ws = (char*)d_ws;
  size_t off = 0;
  auto allocF = [&](size_t n) { float* p = (float*)(ws + off);
                                off += ((n * 4 + 255) & ~(size_t)255); return p; };
  auto allocB = [&](size_t n) { bf16_t* p = (bf16_t*)(ws + off);
                                off += ((n * 2 + 255) & ~(size_t)255); return p; };

  float* x      = allocF(NAd * Dn);
  float* xn     = allocF(NAd * Dn);
  float* r      = allocF(NAd * Dn);
  float* k      = allocF(NAd * Dn);
  float* v      = allocF(NAd * Dn);
  float* wraw   = allocF(NAd * Dn);
  float* aB     = allocF(NAd * Dn);
  float* vgate  = allocF(NAd * Dn);
  float* g      = allocF(NAd * Dn);
  float* a4full = allocF(NAd * Dn * 2);   // a4 | b4 contiguous (aliased by logits)
  float* a4 = a4full;
  float* b4 = a4full + NAd * Dn;
  float* vfirst = allocF(NAd * Dn);
  float* oScan  = allocF(NAd * Dn);

  bf16_t* bxr = allocB(NAd * Dn);
  bf16_t* bxw = allocB(NAd * Dn);
  bf16_t* bxk = allocB(NAd * Dn);
  bf16_t* bxv = allocB(NAd * Dn);
  bf16_t* bxa = allocB(NAd * Dn);
  bf16_t* bxg = allocB(NAd * Dn);
  bf16_t* lr  = allocB((size_t)Dn * MT * 64);
  bf16_t* kc  = allocB(NAd * Dn);
  bf16_t* kk2 = allocB((size_t)Dn * MT * 4 * Hd);
  bf16_t* sbf = allocB(NAd * Dn);   // shared bf16 scratch: zbf / og / hbf

  // bf16 transposed weights
  const long WSZ = (long)Hd * Hd, LR1 = (long)Hd * 32, LR2 = 32L * Hd;
  const long GG1 = (long)Hd * 64, GG2 = 64L * Hd, CW = (long)Hd * 4 * Hd;
  bf16_t* Wr_t  = allocB(12 * WSZ);
  bf16_t* Wk_t  = allocB(12 * WSZ);
  bf16_t* Wv_t  = allocB(12 * WSZ);
  bf16_t* Wo_t  = allocB(12 * WSZ);
  bf16_t* w1_t  = allocB(12 * LR1);
  bf16_t* w2_t  = allocB(12 * LR2);
  bf16_t* a1_t  = allocB(12 * LR1);
  bf16_t* a2_t  = allocB(12 * LR2);
  bf16_t* v1_t  = allocB(12 * LR1);
  bf16_t* v2_t  = allocB(12 * LR2);
  bf16_t* g1_t  = allocB(12 * GG1);
  bf16_t* g2_t  = allocB(12 * GG2);
  bf16_t* cWk_t = allocB(12 * CW);
  bf16_t* cWv_t = allocB(12 * CW);
  bf16_t* down_t = allocB((long)Cc * Hd);
  bf16_t* up_t   = allocB((long)Hd * Cc);
  bf16_t* h1W1_t = allocB((long)Cc * 256);
  bf16_t* h1W2_t = allocB(256L * Cc);
  bf16_t* h2W1_t = allocB((long)Cc * 256);
  bf16_t* h2W2_t = allocB(256L * Cc);
  bf16_t* hm_t   = allocB((long)Cc * 2 * Cc);

  // head-phase aliases (core buffers are free by then)
  float*  hC      = k;
  bf16_t* lnhB    = (bf16_t*)bxr;       // MT*Cc bf16 fits in NAd*Dn slot
  bf16_t* zsnk    = (bf16_t*)bxk;       // MT*256 bf16
  float*  r1buf   = wraw;
  float*  r2buf   = aB;
  float*  logitsF = a4full;             // MT*2C floats

  // --- weight conversion (f32 (K,N) -> bf16 (N,K)) ---
  auto convT = [&](const float* src, bf16_t* dst, int K, int N, int batch) {
    long tot = (long)K * N;
    int gx = (int)((tot + 255) / 256); if (gx > 2048) gx = 2048;
    dim3 gg(gx, batch);
    conv_tr_kernel<<<gg, 256, 0, stream>>>(src, dst, K, N);
  };
  convT(IN_F(C_W_r), Wr_t, Hd, Hd, 12);
  convT(IN_F(C_W_k), Wk_t, Hd, Hd, 12);
  convT(IN_F(C_W_v), Wv_t, Hd, Hd, 12);
  convT(IN_F(C_W_o), Wo_t, Hd, Hd, 12);
  convT(IN_F(C_w1), w1_t, Hd, 32, 12);
  convT(IN_F(C_w2), w2_t, 32, Hd, 12);
  convT(IN_F(C_a1), a1_t, Hd, 32, 12);
  convT(IN_F(C_a2), a2_t, 32, Hd, 12);
  convT(IN_F(C_v1), v1_t, Hd, 32, 12);
  convT(IN_F(C_v2), v2_t, 32, Hd, 12);
  convT(IN_F(C_g1), g1_t, Hd, 64, 12);
  convT(IN_F(C_g2), g2_t, 64, Hd, 12);
  convT(IN_F(C_cWk), cWk_t, Hd, 4 * Hd, 12);
  convT(IN_F(C_cWv), cWv_t, 4 * Hd, Hd, 12);
  convT(IN_F(P_down_W), down_t, Cc, Hd, 1);
  convT(IN_F(P_up_W),   up_t,   Hd, Cc, 1);
  convT(IN_F(H1_W1), h1W1_t, Cc, 256, 1);
  convT(IN_F(H1_W2), h1W2_t, 256, Cc, 1);
  convT(IN_F(H2_W1), h2W1_t, Cc, 256, 1);
  convT(IN_F(H2_W2), h2W2_t, 256, Cc, 1);
  convT(IN_F(P_hm_W), hm_t, Cc, 2 * Cc, 1);

  // --- down projection: x = z_mix @ down_W + down_b (written to both dirs) ---
  conv_bf_kernel<<<(int)((MT * Cc + 255) / 256), 256, 0, stream>>>(zmix, sbf, MT * Cc);
  gemm(stream, sbf, down_t, IN_F(P_down_b), x, false, 0, false,
       (int)MT, Hd, Cc, Dn, 0, 0, 0, NAd);

  const long pS = (long)Lc * Hd;   // per-direction stride of (2,L,Hd) params
  const int  DBT = (int)(Dn * MT);

  // --- core layers (both directions batched via grid.z) ---
  for (int l = 0; l < Lc; ++l) {
    const long lv = (long)l * Hd;
    ln_kernel<float><<<DBT, 256, 0, stream>>>(
        x, IN_F(C_ln1_w) + lv, IN_F(C_ln1_b) + lv, xn, Hd, (int)MT, pS, 1e-5f);
    mix6_kernel<<<DBT, 256, 0, stream>>>(
        xn, IN_F(C_x_r) + lv, IN_F(C_x_w) + lv, IN_F(C_x_k) + lv,
        IN_F(C_x_v) + lv, IN_F(C_x_a) + lv, IN_F(C_x_g) + lv, pS,
        bxr, bxw, bxk, bxv, bxa, bxg);

    gemm(stream, bxr, Wr_t + l * WSZ, nullptr, r, false, 0, false,
         (int)MT, Hd, Hd, Dn, NAd, Lc * WSZ, 0, NAd);
    gemm(stream, bxk, Wk_t + l * WSZ, nullptr, k, false, 0, false,
         (int)MT, Hd, Hd, Dn, NAd, Lc * WSZ, 0, NAd);
    gemm(stream, bxv, Wv_t + l * WSZ, nullptr, v, false, 0, false,
         (int)MT, Hd, Hd, Dn, NAd, Lc * WSZ, 0, NAd);

    // w = w0 + tanh(xw@w1)@w2
    gemm(stream, bxw, w1_t + l * LR1, nullptr, lr, true, 2, false,
         (int)MT, 32, Hd, Dn, NAd, Lc * LR1, 0, MT * 32);
    gemm(stream, lr, w2_t + l * LR2, IN_F(C_w0) + lv, wraw, false, 0, false,
         (int)MT, Hd, 32, Dn, MT * 32, Lc * LR2, pS, NAd);
    // a = sigmoid(a0 + (xa@a1)@a2)
    gemm(stream, bxa, a1_t + l * LR1, nullptr, lr, true, 0, false,
         (int)MT, 32, Hd, Dn, NAd, Lc * LR1, 0, MT * 32);
    gemm(stream, lr, a2_t + l * LR2, IN_F(C_a0) + lv, aB, false, 1, false,
         (int)MT, Hd, 32, Dn, MT * 32, Lc * LR2, pS, NAd);
    // vgate = sigmoid(v0 + (xv@v1)@v2)
    gemm(stream, bxv, v1_t + l * LR1, nullptr, lr, true, 0, false,
         (int)MT, 32, Hd, Dn, NAd, Lc * LR1, 0, MT * 32);
    gemm(stream, lr, v2_t + l * LR2, IN_F(C_v0) + lv, vgate, false, 1, false,
         (int)MT, Hd, 32, Dn, MT * 32, Lc * LR2, pS, NAd);
    // g = sigmoid(xg@g1)@g2
    gemm(stream, bxg, g1_t + l * GG1, nullptr, lr, true, 1, false,
         (int)MT, 64, Hd, Dn, NAd, Lc * GG1, 0, MT * 64);
    gemm(stream, lr, g2_t + l * GG2, nullptr, g, false, 0, false,
         (int)MT, Hd, 64, Dn, MT * 64, Lc * GG2, 0, NAd);

    vecops_kernel<<<DBT, 256, 0, stream>>>(
        k, v, aB, wraw, vgate, vfirst, a4, b4,
        IN_F(C_k_k) + lv, IN_F(C_k_a) + lv, pS, l);

    wkv_scan_kernel<<<Dn * Bc * nHd, 64, 0, stream>>>(r, wraw, k, v, a4, b4, oScan);

    postgn_kernel<<<DBT, 256, 0, stream>>>(
        oScan, r, k, v, g, IN_F(C_lnx_w) + lv, IN_F(C_lnx_b) + lv,
        IN_F(C_r_k) + lv, pS, sbf);

    gemm(stream, sbf, Wo_t + l * WSZ, nullptr, x, false, 0, true,
         (int)MT, Hd, Hd, Dn, NAd, Lc * WSZ, 0, NAd);

    // channel mix
    ln_kernel<float><<<DBT, 256, 0, stream>>>(
        x, IN_F(C_ln2_w) + lv, IN_F(C_ln2_b) + lv, xn, Hd, (int)MT, pS, 1e-5f);
    cmix_kernel<<<DBT, 256, 0, stream>>>(xn, IN_F(C_c_xk) + lv, pS, kc);
    gemm(stream, kc, cWk_t + l * CW, nullptr, kk2, true, 3, false,
         (int)MT, 4 * Hd, Hd, Dn, NAd, Lc * CW, 0, MT * 4 * Hd);
    gemm(stream, kk2, cWv_t + l * CW, nullptr, x, false, 0, true,
         (int)MT, Hd, 4 * Hd, Dn, MT * 4 * Hd, Lc * CW, 0, NAd);
  }

  // --- heads ---
  hcombine_kernel<<<(int)(NAd / 256), 256, 0, stream>>>(x, sbf);
  gemm(stream, sbf, up_t, IN_F(P_up_b), hC, false, 0, false,
       (int)MT, Cc, Hd, 1, 0, 0, 0, 0);

  ln_kernel<bf16_t><<<(int)MT, 256, 0, stream>>>(
      hC, IN_F(H1_ln_w), IN_F(H1_ln_b), lnhB, Cc, (int)MT, 0, 1e-5f);
  gemm(stream, lnhB, h1W1_t, IN_F(H1_b1), zsnk, true, 4, false,
       (int)MT, 256, Cc, 1, 0, 0, 0, 0, IN_F(H1_alpha));
  gemm(stream, zsnk, h1W2_t, IN_F(H1_b2), r1buf, false, 0, false,
       (int)MT, Cc, 256, 1, 0, 0, 0, 0);

  ln_kernel<bf16_t><<<(int)MT, 256, 0, stream>>>(
      hC, IN_F(H2_ln_w), IN_F(H2_ln_b), lnhB, Cc, (int)MT, 0, 1e-5f);
  gemm(stream, lnhB, h2W1_t, IN_F(H2_b1), zsnk, true, 4, false,
       (int)MT, 256, Cc, 1, 0, 0, 0, 0, IN_F(H2_alpha));
  gemm(stream, zsnk, h2W2_t, IN_F(H2_b2), r2buf, false, 0, false,
       (int)MT, Cc, 256, 1, 0, 0, 0, 0);

  ln_kernel<bf16_t><<<(int)MT, 256, 0, stream>>>(
      hC, IN_F(P_hm_ln_w), IN_F(P_hm_ln_b), lnhB, Cc, (int)MT, 0, 1e-5f);
  gemm(stream, lnhB, hm_t, IN_F(P_hm_b), logitsF, false, 0, false,
       (int)MT, 2 * Cc, Cc, 1, 0, 0, 0, 0);

  final_kernel<<<(int)((MT * Cc) / 256), 256, 0, stream>>>(
      zmix, logitsF, r1buf, r2buf, (float*)d_out);
#undef IN_F
}
